// ESN_64759516889579
// MI455X (gfx1250) — matrix operand order repaired and checked by reading, not compiled, not run on hardware
//
#include <hip/hip_runtime.h>
#include <hip/hip_bf16.h>
#include <math.h>

#define RN 4096      // reservoir size N
#define RD 64        // input dim D
#define RO 64        // output dim O
#define TSTEPS 4096  // timesteps T
#define LEAK 0.8f
#define NB 256       // persistent scan blocks
#define NT 256       // threads per scan block (8 waves)

typedef __attribute__((ext_vector_type(16))) _Float16 v16h;
typedef __attribute__((ext_vector_type(2)))  _Float16 v2h;
typedef __attribute__((ext_vector_type(8)))  float    v8f;

// ---------------------------------------------------------------------------
// Init: W (f32) -> Wh (f16), zero x buffer 0, zero barrier state.
// ---------------------------------------------------------------------------
__global__ void esn_init_kernel(const float* __restrict__ W,
                                _Float16* __restrict__ Wh,
                                float* __restrict__ xbuf,
                                int* __restrict__ syncv) {
  const long long total = (long long)RN * RN;
  for (long long i = (long long)blockIdx.x * blockDim.x + threadIdx.x;
       i < total; i += (long long)gridDim.x * blockDim.x) {
    Wh[i] = (_Float16)W[i];
    if (i < RN) xbuf[i] = 0.0f;
    if (i < 32) syncv[i] = 0;
  }
}

// ---------------------------------------------------------------------------
// Pack B = Wo^T (N x O) into WMMA 16x16x32 f16 B-layout, per ISA 7.12.2:
// element e of lane holds B[k][n], n = lane&15, k = kc*32 + (e>>3)*16 + (lane>>4)*8 + (e&7).
// Bpack[ct][kc][lane][e], ct = output column tile (0..3), kc = k-chunk (0..127).
// ---------------------------------------------------------------------------
__global__ void esn_bpack_kernel(const float* __restrict__ Wout,
                                 _Float16* __restrict__ Bpack) {
  int idx = blockIdx.x * blockDim.x + threadIdx.x;   // 0 .. 262143
  int e    =  idx        & 15;
  int lane = (idx >> 4)  & 31;
  int kc   = (idx >> 9)  & 127;
  int ct   =  idx >> 16;
  int n = ct * 16 + (lane & 15);
  int k = kc * 32 + ((e >> 3) * 16) + ((lane >> 4) * 8) + (e & 7);
  Bpack[idx] = (_Float16)Wout[(size_t)n * (RN + 1) + k];
}

// ---------------------------------------------------------------------------
// Persistent reservoir scan. Grid = NB x NT, rows/block = 16, 2 rows/wave.
// Per step: stage x,u in LDS; wave dot-products (f16 W, f32 acc); shuffle
// reduce; tanh + leak; write x_new to double buffer and to WMMA-packed Apack
// (A[t][r] in 16-bit A-matrix 16x32 layout, ISA 7.12.2); device-wide barrier.
// ---------------------------------------------------------------------------
__global__ void __launch_bounds__(NT) esn_scan_kernel(
    const _Float16* __restrict__ Wh,
    const float* __restrict__ Win,
    const float* __restrict__ bvec,
    const float* __restrict__ inputs,
    float* __restrict__ xbuf,          // 2 * RN floats (double buffer)
    _Float16* __restrict__ Apack,      // T*N halves, WMMA A layout
    int* __restrict__ syncv) {
  __shared__ float xs[RN];
  __shared__ float us[RD];

  const int tid  = threadIdx.x;
  const int lane = tid & 31;
  const int wave = tid >> 5;
  const int r0 = blockIdx.x * 16 + wave * 2;   // even
  const int r1 = r0 + 1;
  const _Float16* w0 = Wh + (size_t)r0 * RN;
  const _Float16* w1 = Wh + (size_t)r1 * RN;
  int* cnt  = syncv;
  int* flag = syncv + 16;

  for (int t = 0; t < TSTEPS; ++t) {
    const float* xcur  = xbuf + (t & 1) * RN;
    float*       xnext = xbuf + ((t + 1) & 1) * RN;

    // stage x (16 KB) and u_t (256 B) into LDS
    for (int i = tid; i < RN / 4; i += NT)
      ((float4*)xs)[i] = ((const float4*)xcur)[i];
    if (tid < RD) us[tid] = inputs[(size_t)t * RD + tid];
    __syncthreads();

    float a0 = 0.0f, a1 = 0.0f;
    for (int k = lane * 2; k < RN; k += 64) {      // coalesced 128B/wave
      float2 xv = *(const float2*)(xs + k);
      v2h wv0 = *(const v2h*)(w0 + k);
      v2h wv1 = *(const v2h*)(w1 + k);
      a0 = fmaf((float)wv0.x, xv.x, a0);
      a0 = fmaf((float)wv0.y, xv.y, a0);
      a1 = fmaf((float)wv1.x, xv.x, a1);
      a1 = fmaf((float)wv1.y, xv.y, a1);
    }
    // W_in @ u contribution (64 MACs spread over 32 lanes)
    a0 = fmaf(Win[(size_t)r0 * RD + lane],      us[lane],      a0);
    a0 = fmaf(Win[(size_t)r0 * RD + lane + 32], us[lane + 32], a0);
    a1 = fmaf(Win[(size_t)r1 * RD + lane],      us[lane],      a1);
    a1 = fmaf(Win[(size_t)r1 * RD + lane + 32], us[lane + 32], a1);

    // wave32 reduction
    for (int off = 16; off > 0; off >>= 1) {
      a0 += __shfl_xor(a0, off, 32);
      a1 += __shfl_xor(a1, off, 32);
    }

    if (lane == 0) {
      float pre0 = a0 + bvec[r0];
      float pre1 = a1 + bvec[r1];
      float xn0 = (1.0f - LEAK) * xs[r0] + LEAK * tanhf(pre0);
      float xn1 = (1.0f - LEAK) * xs[r1] + LEAK * tanhf(pre1);
      xnext[r0] = xn0;
      xnext[r1] = xn1;
      // scatter into WMMA A-matrix layout: A[t][r]
      int m = t & 15;
      size_t base = ((size_t)(t >> 4) * 128 + (size_t)(r0 >> 5)) * 512;
      int kin0 = r0 & 31, kin1 = r1 & 31;
      int l0 = ((kin0 >> 3) & 1) * 16 + m;
      int e0 = ((kin0 >> 4) & 1) * 8 + (kin0 & 7);
      int l1 = ((kin1 >> 3) & 1) * 16 + m;
      int e1 = ((kin1 >> 4) & 1) * 8 + (kin1 & 7);
      Apack[base + (size_t)l0 * 16 + e0] = (_Float16)xn0;
      Apack[base + (size_t)l1 * 16 + e1] = (_Float16)xn1;
    }

    // ---- device-wide barrier (monotonic counter + release flag) ----
    __syncthreads();
    if (tid == 0) {
      __threadfence();
      int prev = __hip_atomic_fetch_add(cnt, 1, __ATOMIC_ACQ_REL,
                                        __HIP_MEMORY_SCOPE_AGENT);
      if (prev == (t + 1) * NB - 1)
        __hip_atomic_store(flag, t + 1, __ATOMIC_RELEASE,
                           __HIP_MEMORY_SCOPE_AGENT);
      while (__hip_atomic_load(flag, __ATOMIC_ACQUIRE,
                               __HIP_MEMORY_SCOPE_AGENT) < t + 1)
        __builtin_amdgcn_s_sleep(1);
    }
    __syncthreads();
  }
}

// ---------------------------------------------------------------------------
// Readout GEMM: Y (T x O) = Apack (T x N, f16) @ Bpack (N x O, f16) + bo,
// f32 accumulation via v_wmma_f32_16x16x32_f16. One 16x16 tile per wave,
// 4 waves/block cover the 64 output columns, 256 blocks cover T.
// ---------------------------------------------------------------------------
__global__ void __launch_bounds__(128) esn_readout_kernel(
    const _Float16* __restrict__ Apack,
    const _Float16* __restrict__ Bpack,
    const float* __restrict__ Wout,
    float* __restrict__ out) {
  const int lane = threadIdx.x & 31;
  const int ct = threadIdx.x >> 5;   // column tile 0..3
  const int rt = blockIdx.x;         // row (time) tile 0..255

  const v16h* A = (const v16h*)Apack;
  const v16h* B = (const v16h*)Bpack;
  size_t abase = (size_t)rt * 128 * 32 + lane;
  size_t bbase = (size_t)ct * 128 * 32 + lane;

  v8f c = {};
  for (int kc = 0; kc < 128; ++kc) {
    v16h a  = A[abase + (size_t)kc * 32];
    v16h bf = B[bbase + (size_t)kc * 32];
    c = __builtin_amdgcn_wmma_f32_16x16x32_f16(
        /*neg_a=*/false, a, /*neg_b=*/false, bf,
        /*c_mod=*/(short)0, c, /*reuse_a=*/false, /*reuse_b=*/false);
  }

  int n  = ct * 16 + (lane & 15);
  float bo = Wout[(size_t)n * (RN + 1) + RN];   // last column of W_out
  int mbase = (lane >> 4) * 8;
  for (int r = 0; r < 8; ++r) {
    int t = rt * 16 + mbase + r;
    out[(size_t)t * RO + n] = c[r] + bo;
  }
}

// ---------------------------------------------------------------------------
extern "C" void kernel_launch(void* const* d_in, const int* in_sizes, int n_in,
                              void* d_out, int out_size, void* d_ws, size_t ws_size,
                              hipStream_t stream) {
  (void)in_sizes; (void)n_in; (void)out_size; (void)ws_size;
  const float* inputs = (const float*)d_in[0];  // (T, D)
  const float* W      = (const float*)d_in[1];  // (N, N)
  const float* Win    = (const float*)d_in[2];  // (N, D)
  const float* bvec   = (const float*)d_in[3];  // (N,)
  const float* Wout   = (const float*)d_in[4];  // (O, N+1)
  float* out = (float*)d_out;                   // (T, O) f32

  char* ws = (char*)d_ws;
  size_t off = 0;
  _Float16* Wh    = (_Float16*)(ws + off); off += (size_t)RN * RN * 2;        // 32 MB
  _Float16* Apack = (_Float16*)(ws + off); off += (size_t)TSTEPS * RN * 2;    // 32 MB
  _Float16* Bpack = (_Float16*)(ws + off); off += (size_t)4 * 128 * 512 * 2;  // 512 KB
  float*    xbuf  = (float*)   (ws + off); off += (size_t)2 * RN * 4;         // 32 KB
  int*      syncv = (int*)     (ws + off); off += 128;

  esn_init_kernel<<<4096, 256, 0, stream>>>(W, Wh, xbuf, syncv);
  esn_bpack_kernel<<<1024, 256, 0, stream>>>(Wout, Bpack);
  esn_scan_kernel<<<NB, NT, 0, stream>>>(Wh, Win, bvec, inputs, xbuf, Apack, syncv);
  esn_readout_kernel<<<256, 128, 0, stream>>>(Apack, Bpack, Wout, out);
}
